// SimpleGNN_7292854468774
// MI455X (gfx1250) — matrix operand order repaired and checked by reading, not compiled, not run on hardware
//
#include <hip/hip_runtime.h>

typedef __attribute__((ext_vector_type(16))) _Float16 v16h;
typedef __attribute__((ext_vector_type(8)))  float    v8f;

#define D_DIM 128
#define WT_ELEMS (128 * 128)

// Transpose + convert all 9 weight matrices (Wi, Wm[0..3], Wu[0..3]) from
// f32 row-major [k][n] to f16 n-major [n][k] so B fragments are contiguous.
__global__ void prep_weights_kernel(const float* __restrict__ Wi,
                                    const float* __restrict__ Wm,
                                    const float* __restrict__ Wu,
                                    _Float16* __restrict__ WT) {
    int tid = blockIdx.x * blockDim.x + threadIdx.x;
    if (tid >= 9 * WT_ELEMS) return;
    int w   = tid >> 14;             // which matrix (0..8)
    int idx = tid & (WT_ELEMS - 1);
    int n   = idx >> 7;
    int k   = idx & 127;
    const float* W = (w == 0) ? Wi
                   : (w <= 4) ? (Wm + (size_t)(w - 1) * WT_ELEMS)
                              : (Wu + (size_t)(w - 5) * WT_ELEMS);
    WT[tid] = (_Float16)W[k * D_DIM + n];
}

// Load one 16x128 A stripe as 4 K-tile fragments (f32 -> f16 in regs).
// lanes 0-15: K = 0..15 +32t ; lanes 16-31: K = 16..31 +32t
__device__ __forceinline__ void load_a_frag(const float* __restrict__ xrow,
                                            int hi, v16h a[4]) {
#pragma unroll
    for (int t = 0; t < 4; ++t) {
        const float* px = xrow + t * 32 + hi * 16;
        float lo[8], h8[8];
        *(float4*)(lo + 0) = *(const float4*)(px + 0);
        *(float4*)(lo + 4) = *(const float4*)(px + 4);
        *(float4*)(h8 + 0) = *(const float4*)(px + 8);
        *(float4*)(h8 + 4) = *(const float4*)(px + 12);
#pragma unroll
        for (int i = 0; i < 8; ++i) {
            a[t][i]     = (_Float16)lo[i];
            a[t][8 + i] = (_Float16)h8[i];
        }
    }
}

// Y = relu(X @ W + b)  (ACC==0)   |   Y += relu(X @ W + b)  (ACC==1)
// One wave computes a 32x128 output stripe (two 16-row A tiles). Each B
// fragment loaded from LDS feeds two independent v_wmma_f32_16x16x32_f16
// chains; bias is staged in LDS so the epilogue never waits on loadcnt.
template <int ACC>
__global__ void __launch_bounds__(256)
gnn_gemm_relu_kernel(const float* __restrict__ X,
                     const _Float16* __restrict__ WT,   // 128x128, n-major f16
                     const float* __restrict__ bias,
                     float* __restrict__ Y,
                     int n_pairs) {
    __shared__ _Float16 ldsW[WT_ELEMS];   // 32 KB
    __shared__ float    ldsBias[D_DIM];   // 512 B
    {
        const uint4* src = (const uint4*)WT;
        uint4*       dst = (uint4*)ldsW;
        const int n16 = (WT_ELEMS * 2) / 16;   // 2048 x 16B
        for (int i = threadIdx.x; i < n16; i += blockDim.x) dst[i] = src[i];
        if (threadIdx.x < D_DIM) ldsBias[threadIdx.x] = bias[threadIdx.x];
    }
    __syncthreads();

    const int wave = threadIdx.x >> 5;
    const int lane = threadIdx.x & 31;
    const int pair = blockIdx.x * (blockDim.x >> 5) + wave;
    if (pair >= n_pairs) return;

    const int hi = lane >> 4;       // half-wave id (0/1)
    const int m  = lane & 15;       // row within tile for A
    const int nl = lane & 15;       // col within N-tile for B/C/D
    const size_t row0 = (size_t)pair * 32;    // 32-row stripe base

    v16h a0[4], a1[4];
    load_a_frag(X + (row0 + m) * D_DIM,      hi, a0);
    load_a_frag(X + (row0 + 16 + m) * D_DIM, hi, a1);

#pragma unroll 1
    for (int j = 0; j < 8; ++j) {
        // B fragments for this N-tile: lanes 0-15 K=0..15, lanes 16-31
        // K=16..31 (32B contiguous per fragment). One LDS clause; bias rides
        // along on the same dscnt wait.
        v16h b[4];
        const _Float16* wcol = ldsW + (j * 16 + nl) * D_DIM;
#pragma unroll
        for (int t = 0; t < 4; ++t)
            b[t] = *(const v16h*)(wcol + t * 32 + hi * 16);
        const float bn = ldsBias[j * 16 + nl];

        v8f acc0 = {0.f, 0.f, 0.f, 0.f, 0.f, 0.f, 0.f, 0.f};
        v8f acc1 = {0.f, 0.f, 0.f, 0.f, 0.f, 0.f, 0.f, 0.f};
#pragma unroll
        for (int t = 0; t < 4; ++t) {
            acc0 = __builtin_amdgcn_wmma_f32_16x16x32_f16(
                false, a0[t], false, b[t], (short)0, acc0, false, false);
            acc1 = __builtin_amdgcn_wmma_f32_16x16x32_f16(
                false, a1[t], false, b[t], (short)0, acc1, false, false);
        }

        float* yp0 = Y + (row0 + (size_t)hi * 8) * D_DIM + j * 16 + nl;
        float* yp1 = yp0 + (size_t)16 * D_DIM;

        if constexpr (ACC) {
            float old0[8], old1[8];
#pragma unroll
            for (int r = 0; r < 8; ++r) {
                old0[r] = yp0[(size_t)r * D_DIM];
                old1[r] = yp1[(size_t)r * D_DIM];
            }
#pragma unroll
            for (int r = 0; r < 8; ++r) {
                float v0 = acc0[r] + bn; v0 = v0 > 0.f ? v0 : 0.f;
                float v1 = acc1[r] + bn; v1 = v1 > 0.f ? v1 : 0.f;
                yp0[(size_t)r * D_DIM] = old0[r] + v0;
                yp1[(size_t)r * D_DIM] = old1[r] + v1;
            }
        } else {
#pragma unroll
            for (int r = 0; r < 8; ++r) {
                float v0 = acc0[r] + bn; v0 = v0 > 0.f ? v0 : 0.f;
                float v1 = acc1[r] + bn; v1 = v1 > 0.f ? v1 : 0.f;
                yp0[(size_t)r * D_DIM] = v0;
                yp1[(size_t)r * D_DIM] = v1;
            }
        }
    }
}

__global__ void zero_kernel(float4* __restrict__ p, int n4) {
    int i = blockIdx.x * blockDim.x + threadIdx.x;
    if (i < n4) p[i] = make_float4(0.f, 0.f, 0.f, 0.f);
}

// One wave per edge: gather message[src] (float4/lane), atomic-add into agg[dst].
__global__ void edge_scatter_kernel(const float* __restrict__ msg,
                                    const long long* __restrict__ src,
                                    const long long* __restrict__ dst,
                                    float* __restrict__ agg, int n_edges) {
    int gid  = blockIdx.x * blockDim.x + threadIdx.x;
    int e    = gid >> 5;
    int lane = gid & 31;
    if (e >= n_edges) return;
    long long s = src[e];
    long long d = dst[e];
    float4 mv = *(const float4*)(msg + s * D_DIM + lane * 4);
    float* ap = agg + d * D_DIM + lane * 4;
    atomicAdd(ap + 0, mv.x);
    atomicAdd(ap + 1, mv.y);
    atomicAdd(ap + 2, mv.z);
    atomicAdd(ap + 3, mv.w);
}

extern "C" void kernel_launch(void* const* d_in, const int* in_sizes, int n_in,
                              void* d_out, int out_size, void* d_ws, size_t ws_size,
                              hipStream_t stream) {
    const float*     x  = (const float*)d_in[0];
    const long long* ei = (const long long*)d_in[1];   // [2, E] int64
    const float*     Wi = (const float*)d_in[2];
    const float*     bi = (const float*)d_in[3];
    const float*     Wm = (const float*)d_in[4];
    const float*     bm = (const float*)d_in[5];
    const float*     Wu = (const float*)d_in[6];
    const float*     bu = (const float*)d_in[7];

    const int N = in_sizes[0] / D_DIM;   // 100000 (multiple of 32)
    const int E = in_sizes[1] / 2;       // 640000
    const int n_pairs = N / 32;          // 3125 32-row stripes

    float* state = (float*)d_out;                        // final output = state
    _Float16* WT = (_Float16*)d_ws;                      // 9 * 32KB f16 weights
    const size_t wt_bytes = (size_t)9 * WT_ELEMS * sizeof(_Float16);
    float* msg = (float*)((char*)d_ws + wt_bytes);
    float* agg = msg + (size_t)N * D_DIM;

    prep_weights_kernel<<<(9 * WT_ELEMS + 255) / 256, 256, 0, stream>>>(Wi, Wm, Wu, WT);

    const int gemm_blocks = (n_pairs + 7) / 8;
    // state = relu(x @ Wi + bi)
    gnn_gemm_relu_kernel<0><<<gemm_blocks, 256, 0, stream>>>(x, WT, bi, state, n_pairs);

    const int n4 = (N * D_DIM) / 4;
    const int zero_blocks = (n4 + 255) / 256;
    const int edge_blocks = (int)(((long long)E * 32 + 255) / 256);

    for (int r = 0; r < 4; ++r) {
        // message = relu(state @ Wm_r + bm_r)
        gnn_gemm_relu_kernel<0><<<gemm_blocks, 256, 0, stream>>>(
            state, WT + (size_t)(1 + r) * WT_ELEMS, bm + r * D_DIM, msg, n_pairs);
        // aggregated = 0
        zero_kernel<<<zero_blocks, 256, 0, stream>>>((float4*)agg, n4);
        // aggregated[dst] += message[src]
        edge_scatter_kernel<<<edge_blocks, 256, 0, stream>>>(msg, ei, ei + E, agg, E);
        // state += relu(aggregated @ Wu_r + bu_r)
        gnn_gemm_relu_kernel<1><<<gemm_blocks, 256, 0, stream>>>(
            agg, WT + (size_t)(5 + r) * WT_ELEMS, bu + r * D_DIM, state, n_pairs);
    }
}
